// PerceiverAttention_35751307772207
// MI455X (gfx1250) — compile-verified, matrix operand order checked by reading, not run on hardware
//
#include <hip/hip_runtime.h>

// ---------------------------------------------------------------------------
// PerceiverAttention for MI455X (gfx1250), bf16 WMMA pipeline, v2.
//  - register double-buffered WMMA GEMMs (overlap global loads with matrix ops)
//  - global_prefetch hints two K-tiles ahead
//  - bf16 attn + split-K attn@v with deterministic partial-sum reduction
// Shapes: b=8, m=4096, n_lat=64, DIM=1024, HEADS=16, DIM_HEAD=64, INNER=1024.
// ---------------------------------------------------------------------------

typedef __attribute__((ext_vector_type(16))) __bf16 v16bf;
typedef __attribute__((ext_vector_type(4)))  __bf16 v4bf;
typedef __attribute__((ext_vector_type(8)))  float  v8f;

#define DIM_      1024
#define HEADS_    16
#define DHEAD_    64
#define INNER_    1024
#define BATCH_    8
#define MEDIA_    4096
#define NLAT_     64
#define KVLEN_    4160           // NLAT_ + MEDIA_
#define SCALE_    0.125f         // 64^-0.5
#define KSPLIT_   5              // 4160 / (5*32) = 26 K-steps per chunk
#define KCHUNK_   (KVLEN_ / KSPLIT_)   // 832

__device__ __forceinline__ v8f wmma_bf16(v16bf a, v16bf b, v8f c) {
  // 8 args: (neg_a, A, neg_b, B, c_mod, C, reuse_a, reuse_b)
  return __builtin_amdgcn_wmma_f32_16x16x32_bf16(false, a, false, b,
                                                 (short)0, c, false, false);
}

__device__ __forceinline__ float warp_sum(float v) {
#pragma unroll
  for (int o = 16; o > 0; o >>= 1) v += __shfl_xor(v, o, 32);
  return v;
}
__device__ __forceinline__ float warp_max(float v) {
#pragma unroll
  for (int o = 16; o > 0; o >>= 1) v = fmaxf(v, __shfl_xor(v, o, 32));
  return v;
}

// ---------------------------------------------------------------------------
// LayerNorm one 1024-wide row per block -> bf16 output.
// ---------------------------------------------------------------------------
__global__ __launch_bounds__(256)
void layernorm_bf16_kernel(const float* __restrict__ in,
                           const float* __restrict__ gamma,
                           const float* __restrict__ beta,
                           __bf16* __restrict__ out) {
  const int row  = blockIdx.x;
  const int tid  = threadIdx.x;
  const int lane = tid & 31, wave = tid >> 5;
  const float4 v = *(const float4*)(in + (long)row * DIM_ + tid * 4);

  float s  = v.x + v.y + v.z + v.w;
  float sq = v.x * v.x + v.y * v.y + v.z * v.z + v.w * v.w;
  s  = warp_sum(s);
  sq = warp_sum(sq);

  __shared__ float shs[8], shq[8], mb[2];
  if (lane == 0) { shs[wave] = s; shq[wave] = sq; }
  __syncthreads();
  if (tid == 0) {
    float S = 0.f, Q = 0.f;
#pragma unroll
    for (int i = 0; i < 8; ++i) { S += shs[i]; Q += shq[i]; }
    const float mu  = S * (1.0f / DIM_);
    const float var = Q * (1.0f / DIM_) - mu * mu;
    mb[0] = mu;
    mb[1] = rsqrtf(var + 1e-5f);
  }
  __syncthreads();
  const float mu = mb[0], rstd = mb[1];

  const int c = tid * 4;
  const float4 g = *(const float4*)(gamma + c);
  const float4 b = *(const float4*)(beta + c);
  v4bf o;
  o.x = (__bf16)((v.x - mu) * rstd * g.x + b.x);
  o.y = (__bf16)((v.y - mu) * rstd * g.y + b.y);
  o.z = (__bf16)((v.z - mu) * rstd * g.z + b.z);
  o.w = (__bf16)((v.w - mu) * rstd * g.w + b.w);
  *(v4bf*)(out + (long)row * DIM_ + c) = o;
}

// ---------------------------------------------------------------------------
// Weight transpose + f32->bf16 cast:  W[K][N] -> Wt[N][K].
// ---------------------------------------------------------------------------
__global__ __launch_bounds__(256)
void transpose_cast_kernel(const float* __restrict__ W,
                           __bf16* __restrict__ Wt, int K, int N) {
  __shared__ float tile[32][33];
  const int n0 = blockIdx.x * 32, k0 = blockIdx.y * 32;
  const int tid = threadIdx.x;
  const int c = tid & 31, r0 = tid >> 5;
#pragma unroll
  for (int i = 0; i < 4; ++i) {
    const int r = r0 + i * 8;
    tile[r][c] = W[(long)(k0 + r) * N + n0 + c];
  }
  __syncthreads();
#pragma unroll
  for (int i = 0; i < 4; ++i) {
    const int nn = r0 + i * 8;
    Wt[(long)(n0 + nn) * K + k0 + c] = (__bf16)tile[c][nn];
  }
}

// ---------------------------------------------------------------------------
// Batched WMMA GEMM, software-pipelined:  C[z] = A[z] @ Bt^T.
// Bt is N-major ([N][K] bf16) so both fragments are contiguous 32B loads.
// Block = 8 waves, tile 64(M) x 128(N); wave: 16M x 64N (4 accumulators).
// Register double buffering: tile kb+32 loads issue before tile kb WMMAs.
// ---------------------------------------------------------------------------
__global__ __launch_bounds__(256)
void gemm_bf16_kernel(const __bf16* __restrict__ A,
                      const __bf16* __restrict__ Bt,
                      void* __restrict__ Cv,
                      int M, int N, int K,
                      long strideA, long strideC, int c_bf16) {
  const int lane = threadIdx.x & 31, wave = threadIdx.x >> 5;
  const int m0 = blockIdx.y * 64 + (wave & 3) * 16;
  const int n0 = blockIdx.x * 128 + (wave >> 2) * 64;
  if (m0 >= M) return;
  const int ln15 = lane & 15;
  const int koff = (lane >> 4) * 16;

  const __bf16* arow = A + (long)blockIdx.z * strideA +
                       (long)(m0 + ln15) * K + koff;
  const __bf16* brow = Bt + (long)(n0 + ln15) * K + koff;
  const long bS = (long)16 * K;

  v8f acc0 = {}, acc1 = {}, acc2 = {}, acc3 = {};

  // Prologue: K-tile 0 fragments.
  v16bf a_c  = *(const v16bf*)(arow);
  v16bf b0_c = *(const v16bf*)(brow);
  v16bf b1_c = *(const v16bf*)(brow + bS);
  v16bf b2_c = *(const v16bf*)(brow + 2 * bS);
  v16bf b3_c = *(const v16bf*)(brow + 3 * bS);

  for (int kb = 32; kb < K; kb += 32) {
    // Issue next tile's loads first; WMMAs below consume the previous tile.
    const v16bf a_n  = *(const v16bf*)(arow + kb);
    const v16bf b0_n = *(const v16bf*)(brow + kb);
    const v16bf b1_n = *(const v16bf*)(brow + bS + kb);
    const v16bf b2_n = *(const v16bf*)(brow + 2 * bS + kb);
    const v16bf b3_n = *(const v16bf*)(brow + 3 * bS + kb);
    __builtin_prefetch(arow + kb + 32, 0, 1);   // global_prefetch two tiles out
    __builtin_prefetch(brow + kb + 32, 0, 1);

    acc0 = wmma_bf16(a_c, b0_c, acc0);
    acc1 = wmma_bf16(a_c, b1_c, acc1);
    acc2 = wmma_bf16(a_c, b2_c, acc2);
    acc3 = wmma_bf16(a_c, b3_c, acc3);

    a_c = a_n; b0_c = b0_n; b1_c = b1_n; b2_c = b2_n; b3_c = b3_n;
  }
  acc0 = wmma_bf16(a_c, b0_c, acc0);
  acc1 = wmma_bf16(a_c, b1_c, acc1);
  acc2 = wmma_bf16(a_c, b2_c, acc2);
  acc3 = wmma_bf16(a_c, b3_c, acc3);

  // f32 16x16 C layout: VGPR r -> M = m0 + r (+8 for lanes 16..31), N = lane&15
  const int rbase = m0 + ((lane >> 4) << 3);
  if (c_bf16) {
    __bf16* C = (__bf16*)Cv + (long)blockIdx.z * strideC;
#pragma unroll
    for (int r = 0; r < 8; ++r) {
      const int i = rbase + r;
      if (i < M) {
        __bf16* cr = C + (long)i * N + n0 + ln15;
        cr[0]  = (__bf16)acc0[r];
        cr[16] = (__bf16)acc1[r];
        cr[32] = (__bf16)acc2[r];
        cr[48] = (__bf16)acc3[r];
      }
    }
  } else {
    float* C = (float*)Cv + (long)blockIdx.z * strideC;
#pragma unroll
    for (int r = 0; r < 8; ++r) {
      const int i = rbase + r;
      if (i < M) {
        float* cr = C + (long)i * N + n0 + ln15;
        cr[0]  = acc0[r];
        cr[16] = acc1[r];
        cr[32] = acc2[r];
        cr[48] = acc3[r];
      }
    }
  }
}

// ---------------------------------------------------------------------------
// Per-head V transpose: vT[b,h,d,j] = kv[b,j,INNER_ + h*64 + d]  (bf16)
// ---------------------------------------------------------------------------
__global__ __launch_bounds__(256)
void vtrans_kernel(const __bf16* __restrict__ kv, __bf16* __restrict__ vT) {
  const int bh = blockIdx.y;
  const int b = bh >> 4, h = bh & 15;
  const int j0 = blockIdx.x * 32;
  __shared__ __bf16 tile[32][72];
  const int tid = threadIdx.x;
  const __bf16* src = kv + (long)b * KVLEN_ * (2 * INNER_) + INNER_ + h * DHEAD_;
#pragma unroll
  for (int p = 0; p < 8; ++p) {
    const int idx = p * 256 + tid;
    const int jl = idx >> 6, d = idx & 63;
    tile[jl][d] = src[(long)(j0 + jl) * (2 * INNER_) + d];
  }
  __syncthreads();
  __bf16* dst = vT + (long)bh * DHEAD_ * KVLEN_;
#pragma unroll
  for (int p = 0; p < 8; ++p) {
    const int idx = p * 256 + tid;
    const int d = idx >> 5, jl = idx & 31;
    dst[(long)d * KVLEN_ + j0 + jl] = tile[jl][d];
  }
}

// ---------------------------------------------------------------------------
// sim[b,h,i,j] = scale * (q_h @ k_h^T), mask==1 -> -inf for media j.
// K=64: preload all 6 fragments, then 4 back-to-back WMMAs (full MLP).
// Block: one (bh, 64-wide j chunk); 8 waves = 4(M=64) x 2(j halves of 32).
// ---------------------------------------------------------------------------
__global__ __launch_bounds__(256)
void sim_kernel(const __bf16* __restrict__ q,
                const __bf16* __restrict__ kv,
                const int* __restrict__ amask,
                float* __restrict__ sim) {
  const int bh = blockIdx.y;
  const int b = bh >> 4, h = bh & 15;
  const int lane = threadIdx.x & 31, wave = threadIdx.x >> 5;
  const int m0 = (wave & 3) * 16;
  const int j0 = blockIdx.x * 64 + (wave >> 2) * 32;
  const int ln15 = lane & 15;
  const int koff = (lane >> 4) * 16;

  const __bf16* arow = q + ((long)b * NLAT_ + m0 + ln15) * INNER_ +
                       h * DHEAD_ + koff;
  const __bf16* bra = kv + (long)b * KVLEN_ * (2 * INNER_) +
                      (long)(j0 + ln15) * (2 * INNER_) + h * DHEAD_ + koff;
  const __bf16* brb = bra + (long)16 * (2 * INNER_);

  // Issue all loads, then all WMMAs.
  const v16bf a0  = *(const v16bf*)(arow);
  const v16bf a1  = *(const v16bf*)(arow + 32);
  const v16bf b00 = *(const v16bf*)(bra);
  const v16bf b01 = *(const v16bf*)(bra + 32);
  const v16bf b10 = *(const v16bf*)(brb);
  const v16bf b11 = *(const v16bf*)(brb + 32);

  v8f acc0 = {}, acc1 = {};
  acc0 = wmma_bf16(a0, b00, acc0);
  acc1 = wmma_bf16(a0, b10, acc1);
  acc0 = wmma_bf16(a1, b01, acc0);
  acc1 = wmma_bf16(a1, b11, acc1);

  float* srow = sim + (long)bh * NLAT_ * KVLEN_;
  const int ja = j0 + ln15, jb = j0 + 16 + ln15;
  const int ma  = (ja >= NLAT_) ? amask[b * MEDIA_ + ja - NLAT_] : 0;
  const int mb2 = (jb >= NLAT_) ? amask[b * MEDIA_ + jb - NLAT_] : 0;
  const int rbase = m0 + ((lane >> 4) << 3);
#pragma unroll
  for (int r = 0; r < 8; ++r) {
    const int i = rbase + r;
    float va = acc0[r] * SCALE_;
    float vb = acc1[r] * SCALE_;
    if (ma == 1) va = -__builtin_inff();
    if (mb2 == 1) vb = -__builtin_inff();
    srow[(long)i * KVLEN_ + ja] = va;
    srow[(long)i * KVLEN_ + jb] = vb;
  }
}

// ---------------------------------------------------------------------------
// Stable softmax over one 4160-long row; reads f32 logits, writes bf16 attn.
// ---------------------------------------------------------------------------
__global__ __launch_bounds__(256)
void softmax_kernel(const float* __restrict__ sim, __bf16* __restrict__ attn) {
  const float* p = sim + (long)blockIdx.x * KVLEN_;
  __bf16* q = attn + (long)blockIdx.x * KVLEN_;
  const int tid = threadIdx.x;
  const int lane = tid & 31, wave = tid >> 5;
  float rv[17];
  float mx = -__builtin_inff();
#pragma unroll
  for (int k = 0; k < 17; ++k) {
    const int j = tid + k * 256;
    if (j < KVLEN_) { rv[k] = p[j]; mx = fmaxf(mx, rv[k]); }
  }
  __shared__ float sh[8];
  mx = warp_max(mx);
  if (lane == 0) sh[wave] = mx;
  __syncthreads();
#pragma unroll
  for (int i = 0; i < 8; ++i) mx = fmaxf(mx, sh[i]);
  __syncthreads();

  float s = 0.f;
#pragma unroll
  for (int k = 0; k < 17; ++k) {
    const int j = tid + k * 256;
    if (j < KVLEN_) { rv[k] = __expf(rv[k] - mx); s += rv[k]; }
  }
  s = warp_sum(s);
  if (lane == 0) sh[wave] = s;
  __syncthreads();
  float tot = 0.f;
#pragma unroll
  for (int i = 0; i < 8; ++i) tot += sh[i];
  const float inv = 1.0f / tot;
#pragma unroll
  for (int k = 0; k < 17; ++k) {
    const int j = tid + k * 256;
    if (j < KVLEN_) q[j] = (__bf16)(rv[k] * inv);
  }
}

// ---------------------------------------------------------------------------
// Split-K attn @ v per (b,h): grid (KSPLIT_, bh). Each block handles 832 of
// the 4160 K(j) positions (26 WMMA K-steps), double-buffered, and writes f32
// partial sums part[kc][b*64+i][h*64+d]. Deterministic (no atomics).
// Block: 8 waves = 4(M=64) x 2(N halves of 32).
// ---------------------------------------------------------------------------
__global__ __launch_bounds__(256)
void av_kernel(const __bf16* __restrict__ attn,
               const __bf16* __restrict__ vT,
               float* __restrict__ part) {
  const int kc = blockIdx.x;
  const int bh = blockIdx.y;
  const int b = bh >> 4, h = bh & 15;
  const int lane = threadIdx.x & 31, wave = threadIdx.x >> 5;
  const int m0 = (wave & 3) * 16;
  const int n0 = (wave >> 2) * 32;
  const int ln15 = lane & 15;
  const int koff = (lane >> 4) * 16;
  const int kbeg = kc * KCHUNK_;

  const __bf16* ar  = attn + ((long)bh * NLAT_ + m0 + ln15) * KVLEN_ + kbeg + koff;
  const __bf16* b0r = vT + ((long)bh * DHEAD_ + n0 + ln15) * KVLEN_ + kbeg + koff;
  const __bf16* b1r = b0r + (long)16 * KVLEN_;

  v8f acc0 = {}, acc1 = {};
  v16bf a_c  = *(const v16bf*)(ar);
  v16bf b0_c = *(const v16bf*)(b0r);
  v16bf b1_c = *(const v16bf*)(b1r);
  for (int kb = 32; kb < KCHUNK_; kb += 32) {
    const v16bf a_n  = *(const v16bf*)(ar + kb);
    const v16bf b0_n = *(const v16bf*)(b0r + kb);
    const v16bf b1_n = *(const v16bf*)(b1r + kb);
    __builtin_prefetch(ar + kb + 32, 0, 1);
    __builtin_prefetch(b0r + kb + 32, 0, 1);
    acc0 = wmma_bf16(a_c, b0_c, acc0);
    acc1 = wmma_bf16(a_c, b1_c, acc1);
    a_c = a_n; b0_c = b0_n; b1_c = b1_n;
  }
  acc0 = wmma_bf16(a_c, b0_c, acc0);
  acc1 = wmma_bf16(a_c, b1_c, acc1);

  const int rbase = m0 + ((lane >> 4) << 3);
#pragma unroll
  for (int r = 0; r < 8; ++r) {
    const int i = rbase + r;
    float* pr = part + ((long)kc * (BATCH_ * NLAT_) + b * NLAT_ + i) * INNER_ +
                h * DHEAD_ + n0 + ln15;
    pr[0]  = acc0[r];
    pr[16] = acc1[r];
  }
}

// ---------------------------------------------------------------------------
// Deterministic reduction of KSPLIT_ partials -> bf16 o[b,i,h*64+d].
// ---------------------------------------------------------------------------
__global__ __launch_bounds__(256)
void reduce_kernel(const float* __restrict__ part, __bf16* __restrict__ ob) {
  const int idx = blockIdx.x * 256 + threadIdx.x;   // 0 .. 524287
  float s = 0.f;
#pragma unroll
  for (int c = 0; c < KSPLIT_; ++c)
    s += part[(long)c * (BATCH_ * NLAT_ * INNER_) + idx];
  ob[idx] = (__bf16)s;
}

// ---------------------------------------------------------------------------
// Host-side launch.
// ---------------------------------------------------------------------------
extern "C" void kernel_launch(void* const* d_in, const int* in_sizes, int n_in,
                              void* d_out, int out_size, void* d_ws, size_t ws_size,
                              hipStream_t stream) {
  const float* x       = (const float*)d_in[0];
  const float* latents = (const float*)d_in[1];
  const int*   amask   = (const int*)d_in[2];
  const float* gamma_m = (const float*)d_in[3];
  const float* beta_m  = (const float*)d_in[4];
  const float* gamma_l = (const float*)d_in[5];
  const float* beta_l  = (const float*)d_in[6];
  const float* Wq      = (const float*)d_in[7];
  const float* Wkv     = (const float*)d_in[8];
  const float* Wout    = (const float*)d_in[9];
  float* out = (float*)d_out;

  // Workspace carve-up (256B aligned).
  char* ws = (char*)d_ws;
  auto carve = [&](size_t bytes) -> void* {
    void* p = (void*)ws;
    ws += (bytes + 255) & ~(size_t)255;
    return p;
  };
  __bf16* xn    = (__bf16*)carve((size_t)BATCH_ * MEDIA_ * DIM_ * 2);       // 64 MB
  __bf16* lnl   = (__bf16*)carve((size_t)BATCH_ * NLAT_ * DIM_ * 2);        // 1 MB
  __bf16* WqT   = (__bf16*)carve((size_t)INNER_ * DIM_ * 2);                // 2 MB
  __bf16* WkvT  = (__bf16*)carve((size_t)2 * INNER_ * DIM_ * 2);            // 4 MB
  __bf16* WoutT = (__bf16*)carve((size_t)DIM_ * INNER_ * 2);                // 2 MB
  __bf16* qb    = (__bf16*)carve((size_t)BATCH_ * NLAT_ * INNER_ * 2);      // 1 MB
  __bf16* kvb   = (__bf16*)carve((size_t)BATCH_ * KVLEN_ * 2 * INNER_ * 2); // 130 MB
  __bf16* vT    = (__bf16*)carve((size_t)BATCH_ * HEADS_ * DHEAD_ * KVLEN_ * 2); // 65 MB
  float*  simb  = (float*) carve((size_t)BATCH_ * HEADS_ * NLAT_ * KVLEN_ * 4);  // 130 MB
  __bf16* abf   = (__bf16*)carve((size_t)BATCH_ * HEADS_ * NLAT_ * KVLEN_ * 2);  // 65 MB
  float*  part  = (float*) carve((size_t)KSPLIT_ * BATCH_ * NLAT_ * INNER_ * 4); // 10 MB
  __bf16* ob    = (__bf16*)carve((size_t)BATCH_ * NLAT_ * INNER_ * 2);      // 1 MB

  // 1) LayerNorms -> bf16
  layernorm_bf16_kernel<<<BATCH_ * MEDIA_, 256, 0, stream>>>(x, gamma_m, beta_m, xn);
  layernorm_bf16_kernel<<<BATCH_ * NLAT_, 256, 0, stream>>>(latents, gamma_l, beta_l, lnl);

  // 2) Weights -> N-major bf16
  transpose_cast_kernel<<<dim3(32, 32), 256, 0, stream>>>(Wq,   WqT,   DIM_, INNER_);
  transpose_cast_kernel<<<dim3(64, 32), 256, 0, stream>>>(Wkv,  WkvT,  DIM_, 2 * INNER_);
  transpose_cast_kernel<<<dim3(32, 32), 256, 0, stream>>>(Wout, WoutT, INNER_, DIM_);

  // 3) q = ln @ Wq   (per batch: 64x1024x1024) -> bf16
  gemm_bf16_kernel<<<dim3(INNER_ / 128, 1, BATCH_), 256, 0, stream>>>(
      lnl, WqT, qb, NLAT_, INNER_, DIM_,
      (long)NLAT_ * DIM_, (long)NLAT_ * INNER_, 1);

  // 4) kv rows 0..63 from ln, rows 64..4159 from xn  -> bf16 [b,4160,2048]
  gemm_bf16_kernel<<<dim3(2 * INNER_ / 128, 1, BATCH_), 256, 0, stream>>>(
      lnl, WkvT, kvb, NLAT_, 2 * INNER_, DIM_,
      (long)NLAT_ * DIM_, (long)KVLEN_ * 2 * INNER_, 1);
  gemm_bf16_kernel<<<dim3(2 * INNER_ / 128, MEDIA_ / 64, BATCH_), 256, 0, stream>>>(
      xn, WkvT, kvb + (size_t)NLAT_ * 2 * INNER_, MEDIA_, 2 * INNER_, DIM_,
      (long)MEDIA_ * DIM_, (long)KVLEN_ * 2 * INNER_, 1);

  // 5) Per-head V transpose -> vT[b,h,d,j]
  vtrans_kernel<<<dim3(KVLEN_ / 32, BATCH_ * HEADS_), 256, 0, stream>>>(kvb, vT);

  // 6) sim = scale * q @ k^T with mask (-inf) fused
  sim_kernel<<<dim3(KVLEN_ / 64, BATCH_ * HEADS_), 256, 0, stream>>>(
      qb, kvb, amask, simb);

  // 7) Softmax rows -> bf16 attn
  softmax_kernel<<<BATCH_ * HEADS_ * NLAT_, 256, 0, stream>>>(simb, abf);

  // 8) Split-K attn @ v -> f32 partials, then deterministic reduce -> bf16 o
  av_kernel<<<dim3(KSPLIT_, BATCH_ * HEADS_), 256, 0, stream>>>(abf, vT, part);
  reduce_kernel<<<(BATCH_ * NLAT_ * INNER_) / 256, 256, 0, stream>>>(part, ob);

  // 9) final = o @ Wout  ([512 x 1024] @ [1024 x 1024]) -> f32 d_out
  gemm_bf16_kernel<<<dim3(DIM_ / 128, (BATCH_ * NLAT_) / 64, 1), 256, 0, stream>>>(
      ob, WoutT, out, BATCH_ * NLAT_, DIM_, INNER_, 0L, 0L, 0);
}
